// Model_40218073760115
// MI455X (gfx1250) — matrix operand order repaired and checked by reading, not compiled, not run on hardware
//
#include <hip/hip_runtime.h>
#include <math.h>

// ---------------------------------------------------------------------------
// Types for CDNA5 WMMA (wave32): 16x16x32 bf16 -> f32 accumulate
// ---------------------------------------------------------------------------
typedef __attribute__((ext_vector_type(16))) __bf16 v16bf;
typedef __attribute__((ext_vector_type(8)))  float  v8f;

static __device__ __forceinline__ __bf16 f2bf(float f) {
  unsigned u = __builtin_bit_cast(unsigned, f);
  unsigned r = u + 0x7FFFu + ((u >> 16) & 1u);   // round-to-nearest-even
  unsigned short h = (unsigned short)(r >> 16);
  return __builtin_bit_cast(__bf16, h);
}

// Load 8 consecutive floats (32B, aligned) and convert into fragment[base..base+7]
template<int BASE>
static __device__ __forceinline__ void cvt8(const float* __restrict__ p, v16bf& f) {
  float4 x = *(const float4*)p;
  float4 y = *(const float4*)(p + 4);
  f[BASE + 0] = f2bf(x.x); f[BASE + 1] = f2bf(x.y);
  f[BASE + 2] = f2bf(x.z); f[BASE + 3] = f2bf(x.w);
  f[BASE + 4] = f2bf(y.x); f[BASE + 5] = f2bf(y.y);
  f[BASE + 6] = f2bf(y.z); f[BASE + 7] = f2bf(y.w);
}

// ---------------------------------------------------------------------------
// Generic WMMA GEMM:  Cout[m,n] = act( sum_k A[m,k]*Bx[k,n] + bias[n] )
//   B_IS_NK == true : B stored N x K row-major (i.e. out = A * B^T)
//   B_IS_NK == false: B stored K x N row-major (i.e. out = A * B)
//   MT: number of 16-row m-tiles accumulated per wave (B-fragment reuse)
// One wave computes MT 16x16 output tiles. All dims multiples of 16, K of 32,
// lda/ldb multiples of 16 (for b128 fragment loads).
// Fragment k order (A and B alike): element s of lane half h holds k = 16h + s (wave32, A 16x32 / B 32x16).
// ---------------------------------------------------------------------------
template<bool B_IS_NK, bool DO_TANH, bool HAS_BIAS, int MT>
__global__ void gemm_bf16_wmma(const float* __restrict__ A, int lda,
                               const float* __restrict__ B, int ldb,
                               const float* __restrict__ bias,
                               float* __restrict__ Cout, int ldc,
                               int M, int N, int K)
{
  const int lane  = threadIdx.x;                       // 0..31
  const int n0    = (blockIdx.x * blockDim.y + threadIdx.y) * 16;
  const int m0    = blockIdx.y * (16 * MT);
  if (n0 >= N) return;                                 // uniform per-wave exit

  const int lrow  = lane & 15;                         // 0..15
  const int khalf = lane >> 4;                         // 0..1
  const int n     = n0 + lrow;

  v8f acc[MT] = {};

  const float* Bbase = B_IS_NK ? (B + (long)n * ldb) : (B + (long)n);
  const float* Abase = A + (long)(m0 + lrow) * lda;

  for (int k0 = 0; k0 < K; k0 += 32) {
    // --- B fragment: lanes 0-15 hold K = k0..k0+15, lanes 16-31 hold +16 ---
    v16bf bfr;
    if (B_IS_NK) {
      cvt8<0>(Bbase + k0 + khalf * 16,     bfr);
      cvt8<8>(Bbase + k0 + khalf * 16 + 8, bfr);
    } else {
#pragma unroll
      for (int e = 0; e < 16; ++e)
        bfr[e] = f2bf(Bbase[(long)(k0 + khalf * 16 + e) * ldb]);
    }

#pragma unroll
    for (int mt = 0; mt < MT; ++mt) {
      // --- A fragment: lanes 0-15 hold K = k0..k0+15, lanes 16-31 hold +16 ---
      const float* Arow = Abase + (long)(mt * 16) * lda;
      v16bf afr;
      cvt8<0>(Arow + k0 + khalf * 16,     afr);
      cvt8<8>(Arow + k0 + khalf * 16 + 8, afr);
      acc[mt] = __builtin_amdgcn_wmma_f32_16x16x32_bf16(
          false, afr, false, bfr, (short)0, acc[mt], false, false);
    }
  }

  // --- D layout: VGPR r -> M = r + 8*khalf, N = lane&15 ---
  float bv = 0.f;
  if (HAS_BIAS) bv = bias[n];                // one load per lane, reused 8*MT times
  float* Cp = Cout + (long)(m0 + khalf * 8) * ldc + n;
#pragma unroll
  for (int mt = 0; mt < MT; ++mt) {
#pragma unroll
    for (int r = 0; r < 8; ++r) {
      float v = acc[mt][r];
      if (HAS_BIAS) v += bv;
      if (DO_TANH)  v = tanhf(v);
      Cp[(long)(mt * 16 + r) * ldc] = v;
    }
  }
}

// ---------------------------------------------------------------------------
// Embedding gather: out[r, :] = emb[idx[r], :]
// ---------------------------------------------------------------------------
__global__ void gather_rows(const float* __restrict__ emb,
                            const int* __restrict__ idx,
                            float* __restrict__ out, int rows, int D)
{
  int i = blockIdx.x * blockDim.x + threadIdx.x;
  if (i >= rows * D) return;
  int r = i / D, c = i - r * D;
  out[i] = emb[(long)idx[r] * D + c];
}

// ---------------------------------------------------------------------------
// 512x512 transpose (LDS-tiled, conflict-free) so the recurrence streams
// WhhT[k*512+j] coalesced across lanes.
// ---------------------------------------------------------------------------
__global__ void transpose512(const float* __restrict__ W, float* __restrict__ WT)
{
  __shared__ float tile[32][33];
  const int bx = blockIdx.x * 32, by = blockIdx.y * 32;
  const int x = threadIdx.x, y = threadIdx.y;          // 32 x 8
#pragma unroll
  for (int i = 0; i < 32; i += 8)
    tile[y + i][x] = W[(long)(by + y + i) * 512 + bx + x];
  __syncthreads();
#pragma unroll
  for (int i = 0; i < 32; i += 8)
    WT[(long)(bx + y + i) * 512 + by + x] = tile[x][y + i];
}

// ---------------------------------------------------------------------------
// Sequential RNN recurrence (the only truly serial part):
//   h <- tanh(base[t,:] + Whh @ h),   base already holds Wih@x + b
// 1024 threads: two threads per output j, each summing 256 k terms over the
// TRANSPOSED weight matrix (coalesced b32 loads, h[k] broadcast from LDS).
// ---------------------------------------------------------------------------
__global__ void rnn_recurrence(const float* __restrict__ base,   // T x 512
                               const float* __restrict__ WhhT,   // 512 x 512 (transposed)
                               const float* __restrict__ h0,     // 512
                               float* __restrict__ Hout, int hstride, int T)
{
  __shared__ float h[512];
  __shared__ float part[512];
  const int tid  = threadIdx.x;        // 0..1023
  const int j    = tid & 511;
  const int half = tid >> 9;           // 0 or 1
  if (half == 0) h[j] = h0[j];
  __syncthreads();
  const int kbeg = half * 256;
  for (int t = 0; t < T; ++t) {
    float s = 0.f;
#pragma unroll 4
    for (int k = 0; k < 256; ++k) {
      const int kk = kbeg + k;
      s += WhhT[(long)kk * 512 + j] * h[kk];
    }
    if (half) part[j] = s;
    __syncthreads();                   // all reads of h done; partials visible
    if (half == 0) {
      float v = tanhf(base[t * 512 + j] + s + part[j]);
      h[j] = v;
      Hout[(long)t * hstride + j] = v;
    }
    __syncthreads();                   // new h visible before next step
  }
}

// ---------------------------------------------------------------------------
// Row softmax over 128 attention scores (one block of 128 threads per row)
// ---------------------------------------------------------------------------
__global__ void softmax128(const float* __restrict__ S, float* __restrict__ A)
{
  __shared__ float red[128];
  const int t = blockIdx.x, j = threadIdx.x;
  float v = S[t * 128 + j];
  red[j] = v; __syncthreads();
  for (int off = 64; off; off >>= 1) {
    if (j < off) red[j] = fmaxf(red[j], red[j + off]);
    __syncthreads();
  }
  float mx = red[0]; __syncthreads();
  float e = expf(v - mx);
  red[j] = e; __syncthreads();
  for (int off = 64; off; off >>= 1) {
    if (j < off) red[j] += red[j + off];
    __syncthreads();
  }
  A[t * 128 + j] = e / red[0];
}

// ---------------------------------------------------------------------------
// Per-row log-softmax over 50000 logits + pick target: lp[t] = logit[tgt]-lse
// ---------------------------------------------------------------------------
__global__ void row_logsoftmax_pick(const float* __restrict__ logits, int N,
                                    const int* __restrict__ enums,
                                    float* __restrict__ lp)
{
  __shared__ float red[256];
  const int t = blockIdx.x;
  const float* row = logits + (long)t * N;
  float mx = -__builtin_inff();
  for (int i = threadIdx.x; i < N; i += 256) mx = fmaxf(mx, row[i]);
  red[threadIdx.x] = mx; __syncthreads();
  for (int off = 128; off; off >>= 1) {
    if (threadIdx.x < off)
      red[threadIdx.x] = fmaxf(red[threadIdx.x], red[threadIdx.x + off]);
    __syncthreads();
  }
  mx = red[0]; __syncthreads();
  float s = 0.f;
  for (int i = threadIdx.x; i < N; i += 256) s += expf(row[i] - mx);
  red[threadIdx.x] = s; __syncthreads();
  for (int off = 128; off; off >>= 1) {
    if (threadIdx.x < off) red[threadIdx.x] += red[threadIdx.x + off];
    __syncthreads();
  }
  if (threadIdx.x == 0) {
    int tgt = enums[t + 1];
    lp[t] = row[tgt] - mx - logf(red[0]);
  }
}

__global__ void sum128(const float* __restrict__ lp, float* __restrict__ out)
{
  __shared__ float red[128];
  const int j = threadIdx.x;
  red[j] = lp[j]; __syncthreads();
  for (int off = 64; off; off >>= 1) {
    if (j < off) red[j] += red[j + off];
    __syncthreads();
  }
  if (j == 0) out[0] = red[0];
}

// ---------------------------------------------------------------------------
extern "C" void kernel_launch(void* const* d_in, const int* in_sizes, int n_in,
                              void* d_out, int out_size, void* d_ws, size_t ws_size,
                              hipStream_t stream)
{
  (void)in_sizes; (void)n_in; (void)out_size; (void)ws_size;
  const int D = 512, SRC = 128, EV = 50000;
  const int Tdec = 128;   // TGT - 1

  const int*   fnums = (const int*)  d_in[0];
  const int*   enums = (const int*)  d_in[1];
  const float* emb_f = (const float*)d_in[2];
  const float* Wih_e = (const float*)d_in[3];
  const float* Whh_e = (const float*)d_in[4];
  const float* b_e   = (const float*)d_in[5];
  const float* h0_e  = (const float*)d_in[6];
  const float* emb_d = (const float*)d_in[7];
  const float* Wih_d = (const float*)d_in[8];
  const float* Whh_d = (const float*)d_in[9];
  const float* b_d   = (const float*)d_in[10];
  const float* h0_d  = (const float*)d_in[11];
  const float* Wm    = (const float*)d_in[12];
  const float* bm    = (const float*)d_in[13];
  const float* Wo    = (const float*)d_in[14];
  const float* bo    = (const float*)d_in[15];

  float* ws = (float*)d_ws;
  float* Xg     = ws;  ws += SRC  * D;        // gathered encoder inputs
  float* Eg     = ws;  ws += Tdec * D;        // gathered decoder inputs
  float* xWe    = ws;  ws += SRC  * D;        // emb_f[fnums] @ Wih_e^T + b_e
  float* eWd    = ws;  ws += Tdec * D;        // emb_d[inp]   @ Wih_d^T + b_d
  float* fencs  = ws;  ws += SRC  * D;        // encoder states
  float* cat    = ws;  ws += Tdec * 2 * D;    // [ C | Hs ] per decoder step
  float* scores = ws;  ws += Tdec * SRC;
  float* alpha  = ws;  ws += Tdec * SRC;
  float* Mm     = ws;  ws += Tdec * D;        // merged attention output
  float* lp     = ws;  ws += Tdec;
  float* WTe    = ws;  ws += D * D;           // Whh_e transposed
  float* WTd    = ws;  ws += D * D;           // Whh_d transposed
  float* logits = ws;                          // Tdec x EV  (25.6 MB)

  const dim3 blk(32, 4);
  auto gx = [](int N) { return (N / 16 + 3) / 4; };

  // Input embedding gathers + recurrent-weight transposes (independent work)
  gather_rows<<<(SRC  * D + 255) / 256, 256, 0, stream>>>(emb_f, fnums, Xg, SRC,  D);
  gather_rows<<<(Tdec * D + 255) / 256, 256, 0, stream>>>(emb_d, enums, Eg, Tdec, D);
  transpose512<<<dim3(16, 16), dim3(32, 8), 0, stream>>>(Whh_e, WTe);
  transpose512<<<dim3(16, 16), dim3(32, 8), 0, stream>>>(Whh_d, WTd);

  // Batched input projections (WMMA): 128x512x512 each
  gemm_bf16_wmma<true, false, true, 1><<<dim3(gx(D), SRC / 16), blk, 0, stream>>>(
      Xg, D, Wih_e, D, b_e, xWe, D, SRC, D, D);
  gemm_bf16_wmma<true, false, true, 1><<<dim3(gx(D), Tdec / 16), blk, 0, stream>>>(
      Eg, D, Wih_d, D, b_d, eWd, D, Tdec, D, D);

  // Sequential recurrences (encoder -> fencs; decoder -> cat[:,512:1024])
  rnn_recurrence<<<1, 1024, 0, stream>>>(xWe, WTe, h0_e, fencs,   D,     SRC);
  rnn_recurrence<<<1, 1024, 0, stream>>>(eWd, WTd, h0_d, cat + D, 2 * D, Tdec);

  // Attention, batched across all 128 decoder steps:
  // scores = Hs @ fencs^T   (128 x 128, K=512)
  gemm_bf16_wmma<true, false, false, 1><<<dim3(gx(SRC), Tdec / 16), blk, 0, stream>>>(
      cat + D, 2 * D, fencs, D, nullptr, scores, SRC, Tdec, SRC, D);
  softmax128<<<Tdec, 128, 0, stream>>>(scores, alpha);
  // C = alpha @ fencs       (128 x 512, K=128) -> cat[:,0:512]
  gemm_bf16_wmma<false, false, false, 1><<<dim3(gx(D), Tdec / 16), blk, 0, stream>>>(
      alpha, SRC, fencs, D, nullptr, cat, 2 * D, Tdec, D, SRC);
  // M = tanh(cat @ Wm^T + bm)   (128 x 512, K=1024)
  gemm_bf16_wmma<true, true, true, 1><<<dim3(gx(D), Tdec / 16), blk, 0, stream>>>(
      cat, 2 * D, Wm, 2 * D, bm, Mm, D, Tdec, D, 2 * D);

  // Output projection, ONE batched GEMM: 128 x 50000 x 512.
  // MT=8 m-tiles per wave -> each Wo B-fragment loaded once for all 8 row
  // tiles; Wo (102.4 MB) streamed exactly once per pass.
  gemm_bf16_wmma<true, false, true, 8><<<dim3(gx(EV), Tdec / (16 * 8)), blk, 0, stream>>>(
      Mm, D, Wo, D, bo, logits, EV, Tdec, EV, D);

  // log-softmax rows, pick targets, final sum -> scalar
  row_logsoftmax_pick<<<Tdec, 256, 0, stream>>>(logits, EV, enums, lp);
  sum128<<<1, 128, 0, stream>>>(lp, (float*)d_out);
}
